// HGTLayer_33569464386232
// MI455X (gfx1250) — matrix operand order repaired and checked by reading, not compiled, not run on hardware
//
#include <hip/hip_runtime.h>
#include <hip/hip_bf16.h>
#include <math.h>

// ---------------------------------------------------------------------------
// HGT layer for MI455X (gfx1250): bf16 WMMA GEMMs + atomic edge-softmax stages
// ---------------------------------------------------------------------------

typedef __attribute__((ext_vector_type(16))) __bf16 v16bf;
typedef __attribute__((ext_vector_type(8)))  __bf16 v8bf;
typedef __attribute__((ext_vector_type(8)))  float  v8f;

#define IN_DIM  256
#define OUT_DIM 256
#define NHEAD   8
#define DKH     32

#define TILE_M 64
#define TILE_N 256
#define TILE_K 32
#define LDA_SH 40    // 64x32 bf16 A tile, padded row stride (multiple of 8)
#define LDB_SH 264   // 32x256 bf16 B tile, padded row stride (multiple of 8)

// ---- WMMA fragment helpers (layouts per CDNA5 ISA 7.12.2) ------------------
__device__ inline v16bf frag_a_lds(const __bf16* sA, int mbase, int lane) {
    int m  = mbase + (lane & 15);
    int kb = (lane >> 4) * 8;          // lanes 16-31 hold K=8..15 / 24..31
    v8bf lo = *(const v8bf*)&sA[m * LDA_SH + kb];
    v8bf hi = *(const v8bf*)&sA[m * LDA_SH + kb + 16];
    v16bf a;
#pragma unroll
    for (int i = 0; i < 8; ++i) { a[i] = lo[i]; a[i + 8] = hi[i]; }
    return a;
}

__device__ inline v16bf frag_b_lds(const __bf16* sB, int nbase, int lane) {
    // lane%16 = N column; elements in table K order (K=8h..8h+7, 16+8h..16+8h+7)
    const int n  = nbase + (lane & 15);
    const int kb = (lane >> 4) * 8;
    v16bf b;
#pragma unroll
    for (int i = 0; i < 8; ++i) { b[i] = sB[(kb + i) * LDB_SH + n]; b[i + 8] = sB[(kb + 16 + i) * LDB_SH + n]; }
    return b;
}

// ---- ordered-uint encoding for float atomic max ----------------------------
__device__ inline unsigned f2ord(float f) {
    unsigned u = __float_as_uint(f);
    return (u & 0x80000000u) ? ~u : (u | 0x80000000u);
}
__device__ inline float ord2f(unsigned u) {
    return __uint_as_float((u & 0x80000000u) ? (u & 0x7fffffffu) : ~u);
}

// ---------------------------------------------------------------------------
// Generic 256-wide GEMM: C = (A * a_scale) @ B + bias  [+ skip-mix epilogue]
// A: f32 [M x Kdim], B: f32 [Kdim x Ndim] (row-major), converted to bf16 into
// double-buffered LDS tiles. Block tile 64x256, 8 waves, each wave 32x64
// (2x4 WMMA subtiles). One barrier per K-step.
// mode 0: store bf16 (projection path).  mode 1: store f32 with skip mix.
// ---------------------------------------------------------------------------
__global__ __launch_bounds__(256)
void hgt_gemm(const float* __restrict__ A, const float* __restrict__ B,
              const float* __restrict__ bias,
              __bf16* __restrict__ outBf, float* __restrict__ outF,
              const float* __restrict__ resid, const float* __restrict__ skip,
              int M, int Kdim, int Ndim, float a_scale, int mode)
{
    __shared__ __bf16 sA[2][TILE_M * LDA_SH];
    __shared__ __bf16 sB[2][TILE_K * LDB_SH];

    const int tid  = threadIdx.x;
    const int wave = tid >> 5, lane = tid & 31;
    const int m0 = blockIdx.x * TILE_M;
    const int n0 = blockIdx.y * TILE_N;
    const int wm = (wave & 1) * 32;    // wave's M offset in block tile
    const int wn = (wave >> 1) * 64;   // wave's N offset in block tile

    v8f acc[2][4] = {};

    // staging coordinates
    const int ar = tid >> 2;            // A tile row (64 rows, 8 f32/thread)
    const int ac = (tid & 3) * 8;       // A tile col
    const int br = tid >> 3;            // B tile row (32 rows, 32 f32/thread)
    const int bc = (tid & 7) * 32;      // B tile col

    const bool aRowOk = (m0 + ar) < M;
    const int nsteps = Kdim / TILE_K;

    auto loadA = [&](int k0, float4& f0, float4& f1) {
        f0 = make_float4(0, 0, 0, 0); f1 = f0;
        if (aRowOk) {
            const float* s = A + (size_t)(m0 + ar) * Kdim + k0 + ac;
            f0 = *(const float4*)(s);
            f1 = *(const float4*)(s + 4);
        }
    };
    auto storeA = [&](int buf, const float4& f0, const float4& f1) {
        v8bf v;
        v[0] = (__bf16)(f0.x * a_scale); v[1] = (__bf16)(f0.y * a_scale);
        v[2] = (__bf16)(f0.z * a_scale); v[3] = (__bf16)(f0.w * a_scale);
        v[4] = (__bf16)(f1.x * a_scale); v[5] = (__bf16)(f1.y * a_scale);
        v[6] = (__bf16)(f1.z * a_scale); v[7] = (__bf16)(f1.w * a_scale);
        *(v8bf*)&sA[buf][ar * LDA_SH + ac] = v;
    };
    auto loadB = [&](int k0, float4* fb) {
        const float* s = B + (size_t)(k0 + br) * Ndim + n0 + bc;
#pragma unroll
        for (int j = 0; j < 8; ++j) fb[j] = *(const float4*)(s + 4 * j);
    };
    auto storeB = [&](int buf, const float4* fb) {
#pragma unroll
        for (int j = 0; j < 4; ++j) {
            v8bf v;
            v[0] = (__bf16)fb[2*j].x;   v[1] = (__bf16)fb[2*j].y;
            v[2] = (__bf16)fb[2*j].z;   v[3] = (__bf16)fb[2*j].w;
            v[4] = (__bf16)fb[2*j+1].x; v[5] = (__bf16)fb[2*j+1].y;
            v[6] = (__bf16)fb[2*j+1].z; v[7] = (__bf16)fb[2*j+1].w;
            *(v8bf*)&sB[buf][br * LDB_SH + bc + j * 8] = v;
        }
    };

    // prologue: stage tile 0
    {
        float4 a0, a1, fb[8];
        loadA(0, a0, a1);
        loadB(0, fb);
        storeA(0, a0, a1);
        storeB(0, fb);
    }
    __syncthreads();

    for (int step = 0; step < nsteps; ++step) {
        const int cur = step & 1;
        float4 na0, na1, nfb[8];
        if (step + 1 < nsteps) {
            // issue next tile's global loads before the math
            loadA((step + 1) * TILE_K, na0, na1);
            loadB((step + 1) * TILE_K, nfb);
            if (step + 2 < nsteps) {
                if (aRowOk)
                    __builtin_prefetch(&A[(size_t)(m0 + ar) * Kdim + (step + 2) * TILE_K + ac], 0, 3);
                __builtin_prefetch(&B[(size_t)((step + 2) * TILE_K + br) * Ndim + n0 + bc], 0, 3);
            }
        }

        v16bf afrag[2], bfrag[4];
        afrag[0] = frag_a_lds(sA[cur], wm + 0,  lane);
        afrag[1] = frag_a_lds(sA[cur], wm + 16, lane);
#pragma unroll
        for (int j = 0; j < 4; ++j) bfrag[j] = frag_b_lds(sB[cur], wn + j * 16, lane);

#pragma unroll
        for (int i = 0; i < 2; ++i)
#pragma unroll
            for (int j = 0; j < 4; ++j)
                acc[i][j] = __builtin_amdgcn_wmma_f32_16x16x32_bf16(
                    false, afrag[i], false, bfrag[j], (short)0, acc[i][j], false, false);

        if (step + 1 < nsteps) {
            storeA(1 - cur, na0, na1);
            storeB(1 - cur, nfb);
        }
        __syncthreads();
    }

    // epilogue
    const int ln = lane & 15;
    const int lm = (lane >> 4) * 8;
    float alpha = 0.f;
    if (mode == 1) alpha = 1.f / (1.f + __expf(-skip[0]));
#pragma unroll
    for (int i = 0; i < 2; ++i) {
#pragma unroll
        for (int j = 0; j < 4; ++j) {
#pragma unroll
            for (int v = 0; v < 8; ++v) {
                int gm = m0 + wm + i * 16 + v + lm;
                int gn = n0 + wn + j * 16 + ln;
                if (gm < M) {
                    float val = acc[i][j][v] + bias[gn];
                    size_t idx = (size_t)gm * Ndim + gn;
                    if (mode == 0) {
                        outBf[idx] = (__bf16)val;
                    } else {
                        outF[idx] = val * alpha + resid[idx] * (1.f - alpha);
                    }
                }
            }
        }
    }
}

// ---------------------------------------------------------------------------
// Per-relation head transform: out[n, h*32+e] = sum_d X[n, h*32+d] * R[h,d,e]
// One WMMA (K=32) per 16x16 output tile. gridDim.y: 0 -> (Kb,rel_att,kr),
// 1 -> (Vb,rel_msg,vr). Wave w handles head w over 64 rows per block.
// ---------------------------------------------------------------------------
__global__ __launch_bounds__(256)
void hgt_relmm(const __bf16* __restrict__ Kb, const float* __restrict__ Ratt,
               __bf16* __restrict__ krOut,
               const __bf16* __restrict__ Vb, const float* __restrict__ Rmsg,
               __bf16* __restrict__ vrOut, int M)
{
    const __bf16* X = (blockIdx.y == 0) ? Kb : Vb;
    const float*  R = (blockIdx.y == 0) ? Ratt : Rmsg;
    __bf16*       O = (blockIdx.y == 0) ? krOut : vrOut;

    const int wave = threadIdx.x >> 5, lane = threadIdx.x & 31;
    const int head = wave;                       // 8 waves == 8 heads
    const int m0 = blockIdx.x * 64;
    const float* Rh = R + (size_t)head * DKH * DKH;   // 32x32 row-major

    // B fragments: lane%16 = e column, elements over d in table K order
    v16bf bfrag[2];
#pragma unroll
    for (int ns = 0; ns < 2; ++ns) {
        const float* rr = Rh + ((lane >> 4) * 8) * DKH + ns * 16 + (lane & 15);
#pragma unroll
        for (int i = 0; i < 8; ++i) { bfrag[ns][i] = (__bf16)rr[i * DKH]; bfrag[ns][i + 8] = (__bf16)rr[(i + 16) * DKH]; }
    }

    const int ln = lane & 15;
    const int lm = (lane >> 4) * 8;
    const int kb = head * DKH + (lane >> 4) * 8;

#pragma unroll
    for (int mt = 0; mt < 4; ++mt) {
        int m = m0 + mt * 16 + ln;
        v16bf a;
        if (m < M) {
            v8bf lo = *(const v8bf*)&X[(size_t)m * OUT_DIM + kb];
            v8bf hi = *(const v8bf*)&X[(size_t)m * OUT_DIM + kb + 16];
#pragma unroll
            for (int i = 0; i < 8; ++i) { a[i] = lo[i]; a[i + 8] = hi[i]; }
        } else {
#pragma unroll
            for (int i = 0; i < 16; ++i) a[i] = (__bf16)0.f;
        }
#pragma unroll
        for (int ns = 0; ns < 2; ++ns) {
            v8f c = {};
            c = __builtin_amdgcn_wmma_f32_16x16x32_bf16(false, a, false, bfrag[ns], (short)0, c, false, false);
#pragma unroll
            for (int v = 0; v < 8; ++v) {
                int gm = m0 + mt * 16 + v + lm;
                int gn = head * DKH + ns * 16 + ln;
                if (gm < M) O[(size_t)gm * OUT_DIM + gn] = (__bf16)c[v];
            }
        }
    }
}

// ---------------------------------------------------------------------------
// Edge score: one wave32 per edge. lane -> 8 contiguous channels; 4 lanes/head.
// score[e,h] = (Q[dst,h,:] . k[src,h,:]) * pri[h] / sqrt(DK); atomicMax per dst.
// ---------------------------------------------------------------------------
__global__ __launch_bounds__(256)
void hgt_edge_score(const __bf16* __restrict__ Qb, const __bf16* __restrict__ kr,
                    const int* __restrict__ srcI, const int* __restrict__ dstI,
                    const float* __restrict__ pri, float inv_sqrt,
                    float* __restrict__ score, unsigned* __restrict__ mbuf, int E)
{
    const int wave = threadIdx.x >> 5, lane = threadIdx.x & 31;
    const int e = blockIdx.x * 8 + wave;
    if (e >= E) return;
    const int s = srcI[e], d = dstI[e];

    v8bf q = *(const v8bf*)&Qb[(size_t)d * OUT_DIM + lane * 8];
    v8bf k = *(const v8bf*)&kr[(size_t)s * OUT_DIM + lane * 8];
    float p = 0.f;
#pragma unroll
    for (int i = 0; i < 8; ++i) p += (float)q[i] * (float)k[i];
    p += __shfl_xor(p, 1, 32);
    p += __shfl_xor(p, 2, 32);

    const int head = lane >> 2;
    if ((lane & 3) == 0) {
        float sc = p * pri[head] * inv_sqrt;
        score[(size_t)e * NHEAD + head] = sc;
        atomicMax(&mbuf[(size_t)d * NHEAD + head], f2ord(sc));
    }
}

// exp(score - max) and per-dst sum
__global__ __launch_bounds__(256)
void hgt_edge_exp(const int* __restrict__ dstI, float* __restrict__ score,
                  const unsigned* __restrict__ mbuf, float* __restrict__ sbuf, int EH)
{
    int i = blockIdx.x * 256 + threadIdx.x;
    if (i >= EH) return;
    int e = i >> 3, h = i & 7;
    int d = dstI[e];
    float m  = ord2f(mbuf[(size_t)d * NHEAD + h]);
    float ex = __expf(score[i] - m);
    score[i] = ex;
    atomicAdd(&sbuf[(size_t)d * NHEAD + h], ex);
}

// scatter: agg[dst] += v[src] * (e / sum[dst]); one wave per edge
__global__ __launch_bounds__(256)
void hgt_edge_scatter(const __bf16* __restrict__ vr, const int* __restrict__ srcI,
                      const int* __restrict__ dstI, const float* __restrict__ score,
                      const float* __restrict__ sbuf, float* __restrict__ agg, int E)
{
    const int wave = threadIdx.x >> 5, lane = threadIdx.x & 31;
    const int e = blockIdx.x * 8 + wave;
    if (e >= E) return;
    const int s = srcI[e], d = dstI[e];
    const int head = lane >> 2;
    float a = score[(size_t)e * NHEAD + head] / sbuf[(size_t)d * NHEAD + head];
    v8bf v = *(const v8bf*)&vr[(size_t)s * OUT_DIM + lane * 8];
    float* dst = &agg[(size_t)d * OUT_DIM + lane * 8];
#pragma unroll
    for (int i = 0; i < 8; ++i) atomicAdd(&dst[i], (float)v[i] * a);
}

// ---------------------------------------------------------------------------
extern "C" void kernel_launch(void* const* d_in, const int* in_sizes, int n_in,
                              void* d_out, int out_size, void* d_ws, size_t ws_size,
                              hipStream_t stream)
{
    const float* h_A   = (const float*)d_in[0];
    const float* h_B   = (const float*)d_in[1];
    const float* Wk    = (const float*)d_in[2];
    const float* bk    = (const float*)d_in[3];
    const float* Wq    = (const float*)d_in[4];
    const float* bq    = (const float*)d_in[5];
    const float* Wv    = (const float*)d_in[6];
    const float* bv    = (const float*)d_in[7];
    const float* Wa    = (const float*)d_in[8];
    const float* ba    = (const float*)d_in[9];
    const float* ratt  = (const float*)d_in[10];
    const float* rmsg  = (const float*)d_in[11];
    const float* rpri  = (const float*)d_in[12];
    const float* skip  = (const float*)d_in[13];

    const int nA = in_sizes[0] / IN_DIM;
    const int nB = in_sizes[1] / IN_DIM;
    const int E  = in_sizes[14];
    const int N[2] = { nA, nB };
    const float* h[2] = { h_A, h_B };

    // ---- workspace bump allocator (256B aligned) ----
    char* base = (char*)d_ws;
    size_t off = 0;
    auto alloc = [&](size_t bytes) -> void* {
        void* p = base + off;
        off += (bytes + 255) & ~(size_t)255;
        return p;
    };
    __bf16* Kb[2], *Qb[2], *Vb[2];
    for (int t = 0; t < 2; ++t) Kb[t] = (__bf16*)alloc((size_t)N[t] * OUT_DIM * 2);
    for (int t = 0; t < 2; ++t) Qb[t] = (__bf16*)alloc((size_t)N[t] * OUT_DIM * 2);
    for (int t = 0; t < 2; ++t) Vb[t] = (__bf16*)alloc((size_t)N[t] * OUT_DIM * 2);
    int nMax = nA > nB ? nA : nB;
    __bf16* krb = (__bf16*)alloc((size_t)nMax * OUT_DIM * 2);
    __bf16* vrb = (__bf16*)alloc((size_t)nMax * OUT_DIM * 2);
    float* agg[2];
    for (int t = 0; t < 2; ++t) agg[t] = (float*)alloc((size_t)N[t] * OUT_DIM * 4);
    float*    score = (float*)alloc((size_t)E * NHEAD * 4);
    unsigned* mbuf  = (unsigned*)alloc((size_t)nMax * NHEAD * 4);
    float*    sbuf  = (float*)alloc((size_t)nMax * NHEAD * 4);
    (void)ws_size;

    const float inv_sqrt = 0.17677669529663687f; // 1/sqrt(32)

    // zero aggregation buffers
    for (int t = 0; t < 2; ++t)
        hipMemsetAsync(agg[t], 0, (size_t)N[t] * OUT_DIM * 4, stream);

    // K/Q/V projections (bf16 outputs)
    for (int t = 0; t < 2; ++t) {
        dim3 grid((N[t] + TILE_M - 1) / TILE_M, OUT_DIM / TILE_N);
        hgt_gemm<<<grid, 256, 0, stream>>>(h[t], Wk + (size_t)t * IN_DIM * OUT_DIM,
            bk + t * OUT_DIM, Kb[t], nullptr, nullptr, nullptr,
            N[t], IN_DIM, OUT_DIM, 1.f, 0);
        hgt_gemm<<<grid, 256, 0, stream>>>(h[t], Wq + (size_t)t * IN_DIM * OUT_DIM,
            bq + t * OUT_DIM, Qb[t], nullptr, nullptr, nullptr,
            N[t], IN_DIM, OUT_DIM, 1.f, 0);
        hgt_gemm<<<grid, 256, 0, stream>>>(h[t], Wv + (size_t)t * IN_DIM * OUT_DIM,
            bv + t * OUT_DIM, Vb[t], nullptr, nullptr, nullptr,
            N[t], IN_DIM, OUT_DIM, 1.f, 0);
    }

    // per-relation pipeline
    const int REL_SRC[3] = {0, 1, 0};
    const int REL_DST[3] = {1, 0, 0};
    for (int r = 0; r < 3; ++r) {
        const int st = REL_SRC[r], dt = REL_DST[r];
        const int* srcI = (const int*)d_in[14 + 2 * r];
        const int* dstI = (const int*)d_in[15 + 2 * r];

        hipMemsetAsync(mbuf, 0, (size_t)N[dt] * NHEAD * 4, stream); // ord(-inf) < 0x00800000
        hipMemsetAsync(sbuf, 0, (size_t)N[dt] * NHEAD * 4, stream);

        dim3 gR((N[st] + 63) / 64, 2);
        hgt_relmm<<<gR, 256, 0, stream>>>(Kb[st], ratt + (size_t)r * NHEAD * DKH * DKH, krb,
                                          Vb[st], rmsg + (size_t)r * NHEAD * DKH * DKH, vrb,
                                          N[st]);

        hgt_edge_score<<<(E + 7) / 8, 256, 0, stream>>>(Qb[dt], krb, srcI, dstI,
                                                        rpri + r * NHEAD, inv_sqrt,
                                                        score, mbuf, E);
        hgt_edge_exp<<<(E * NHEAD + 255) / 256, 256, 0, stream>>>(dstI, score, mbuf, sbuf,
                                                                  E * NHEAD);
        hgt_edge_scatter<<<(E + 7) / 8, 256, 0, stream>>>(vrb, srcI, dstI, score, sbuf,
                                                          agg[dt], E);
    }

    // output GEMM: new_h[t] = sigmoid(skip)*(mean(agg) @ Wa + ba) + (1-a)*h[t]
    float* outF = (float*)d_out;
    const float relScale[2] = { 0.5f, 1.0f };   // A targeted by rels 1,2; B by rel 0
    for (int t = 0; t < 2; ++t) {
        dim3 grid((N[t] + TILE_M - 1) / TILE_M, OUT_DIM / TILE_N);
        hgt_gemm<<<grid, 256, 0, stream>>>(agg[t], Wa + (size_t)t * OUT_DIM * OUT_DIM,
            ba + t * OUT_DIM, nullptr,
            outF + (t == 0 ? 0 : (size_t)nA * OUT_DIM),
            h[t], skip + t,
            N[t], OUT_DIM, OUT_DIM, relScale[t], 1);
    }
}
